// LocalMultiRangeAttention_2302102471286
// MI455X (gfx1250) — matrix operand order repaired and checked by reading, not compiled, not run on hardware
//
#include <hip/hip_runtime.h>
#include <hip/hip_bf16.h>

typedef _Float16 f16;
typedef __attribute__((ext_vector_type(16))) _Float16 v16h;
typedef __attribute__((ext_vector_type(8)))  float    v8f;

#define HW     4096   // 64*64 pixels per batch image
#define WDIM   64
#define KDEPTH 256    // GEMM reduction depth (input channels)
#define NKB    8      // KDEPTH / 32 K-blocks
#define NPT    256    // HW / 16 pixel tiles per batch
#define NHEADS 8
#define HDIM   32
#define SCALE  0.17677669529663687f  // 32^-0.5

// ---------------------------------------------------------------------------
// Pack W (O x 256 fp32 row-major) -> f16 WMMA A-fragments.
// Wp[mtile][kb][lane][e], 16 halfwords (32B) per (tile,lane):
//   lanes 0-15 : row = mtile*16+lane,    e0..15 = K+0..15
//   lanes 16-31: row = mtile*16+lane-16, e0..15 = K+16..31
// One thread per (mtile, kb, lane). Grid covers exactly (O/16)*8*32 threads.
// ---------------------------------------------------------------------------
__global__ void __launch_bounds__(256)
pack_w_f16(const float* __restrict__ W, f16* __restrict__ Wp)
{
    const int t    = blockIdx.x * 256 + threadIdx.x;
    const int lane = t & 31;
    const int kb   = (t >> 5) & 7;
    const int mt   = t >> 8;
    const int row  = (mt << 4) + (lane & 15);
    const int k0   = (kb << 5) + ((lane & 16) ? 16 : 0);
    const float* wr = W + (size_t)row * KDEPTH + k0;
    v16h v;
#pragma unroll
    for (int e = 0; e < 8; ++e) v[e]     = (f16)wr[e];
#pragma unroll
    for (int e = 0; e < 8; ++e) v[8 + e] = (f16)wr[8 + e];
    *(v16h*)(Wp + (size_t)t * 16) = v;
}

// ---------------------------------------------------------------------------
// Pack X (4, 256, 4096 fp32 channel-major) -> f16 WMMA B-fragments.
// Xp[b][kb][ptile][lane][e], 16 halfwords per (tile,lane):
//   column N = ptile*16 + (lane&15); lanes 0-15: K = kb*32+e, lanes 16-31: +16.
// One thread per (b, kb, ptile, lane) = 262144 threads (grid 1024 x 256).
// This performs the strided fp32 gather + f16 convert exactly ONCE per element.
// ---------------------------------------------------------------------------
__global__ void __launch_bounds__(256)
pack_x_f16(const float* __restrict__ X, f16* __restrict__ Xp)
{
    const int t     = blockIdx.x * 256 + threadIdx.x;
    const int lane  = t & 31;
    const int ptile = (t >> 5) & 255;
    const int kb    = (t >> 13) & 7;
    const int b     = t >> 16;
    const int pix   = (ptile << 4) + (lane & 15);
    const int k0    = (kb << 5) + ((lane & 16) ? 16 : 0);
    const float* xp = X + ((size_t)b * KDEPTH + k0) * HW + pix;
    v16h v;
#pragma unroll
    for (int e = 0; e < 16; ++e) v[e] = (f16)xp[(size_t)e * HW];
    *(v16h*)(Xp + (size_t)t * 16) = v;
}

// ---------------------------------------------------------------------------
// GEMM on packed operands: Y[b,row,p] = sum_c W[row,c]*X[b,c,p] + bias[row].
// One wave -> 16 rows x 64 pixels (4 N-subtiles sharing one A fragment).
// Inner loop per K-block: 1x32B A load + 4x32B B loads + 4 WMMAs.
// Block = 8 waves = 512 pixels. grid = (8, O/16, 4).
// ---------------------------------------------------------------------------
__global__ void __launch_bounds__(256)
gemm_wmma_packed(const f16* __restrict__ Wp, const float* __restrict__ bias,
                 const f16* __restrict__ Xp, float* __restrict__ Y, int ochan)
{
    const int lane  = threadIdx.x & 31;
    const int wv    = threadIdx.x >> 5;
    const int b     = blockIdx.z;
    const int mt    = blockIdx.y;
    const int pt0   = (blockIdx.x * 8 + wv) * 4;   // first of 4 pixel tiles
    const int rowSel = lane & 15;
    const int hi     = (lane >> 4) & 1;

    v8f acc0 = {}, acc1 = {}, acc2 = {}, acc3 = {};

#pragma unroll
    for (int kb = 0; kb < NKB; ++kb) {
        const v16h a = *(const v16h*)(Wp + (((size_t)mt * NKB + kb) * 32 + lane) * 16);
        const f16* xb = Xp + ((((size_t)b * NKB + kb) * NPT + pt0) * 32 + lane) * 16;
        const v16h b0 = *(const v16h*)(xb);
        const v16h b1 = *(const v16h*)(xb + 512);
        const v16h b2 = *(const v16h*)(xb + 1024);
        const v16h b3 = *(const v16h*)(xb + 1536);
        acc0 = __builtin_amdgcn_wmma_f32_16x16x32_f16(false, a, false, b0, (short)0, acc0, false, false);
        acc1 = __builtin_amdgcn_wmma_f32_16x16x32_f16(false, a, false, b1, (short)0, acc1, false, false);
        acc2 = __builtin_amdgcn_wmma_f32_16x16x32_f16(false, a, false, b2, (short)0, acc2, false, false);
        acc3 = __builtin_amdgcn_wmma_f32_16x16x32_f16(false, a, false, b3, (short)0, acc3, false, false);
    }

    // D: VGPR r -> row (r + 8*hi), column rowSel within each 16-pixel subtile
    const int p0 = pt0 << 4;
#pragma unroll
    for (int r = 0; r < 8; ++r) {
        const int   row = (mt << 4) + r + (hi ? 8 : 0);
        const float bv  = bias[row];
        float* yp = Y + ((size_t)b * ochan + row) * HW + p0 + rowSel;
        yp[0]  = acc0[r] + bv;
        yp[16] = acc1[r] + bv;
        yp[32] = acc2[r] + bv;
        yp[48] = acc3[r] + bv;
    }
}

// ---------------------------------------------------------------------------
// Dilated neighborhood attention, online softmax, fp32 VALU.
// One thread per (b, head, pixel); all 32 lanes of a wave share (b, head).
// Zero-padded neighbors participate in softmax with logit = rpb, v = 0.
// ---------------------------------------------------------------------------
__global__ void __launch_bounds__(256)
attn_multirange(const float* __restrict__ qkv,
                const float* __restrict__ rpb0,
                const float* __restrict__ rpb1,
                const float* __restrict__ rpb2,
                float* __restrict__ att)
{
    const int idx  = blockIdx.x * 256 + threadIdx.x;
    const int p    = idx & (HW - 1);
    const int bh   = idx >> 12;
    const int head = bh & (NHEADS - 1);
    const int b    = bh >> 3;
    const int py   = p >> 6;
    const int px   = p & 63;

    int ks, dil;
    const float* rpb;
    if (head < 4)      { ks = 5; dil = 1; rpb = rpb0 + head * 25; }
    else if (head < 7) { ks = 7; dil = 2; rpb = rpb1 + (head - 4) * 49; }
    else               { ks = 9; dil = 3; rpb = rpb2; }

    const size_t hbase = ((size_t)b * 768 + head * HDIM) * HW;
    const float* qp  = qkv + hbase + p;
    const float* kbp = qkv + hbase + (size_t)256 * HW;
    const float* vbp = qkv + hbase + (size_t)512 * HW;

    float q[HDIM];
#pragma unroll
    for (int d = 0; d < HDIM; ++d) q[d] = qp[(size_t)d * HW] * SCALE;

    float m = -3.0e38f, s = 0.0f;
    float o[HDIM];
#pragma unroll
    for (int d = 0; d < HDIM; ++d) o[d] = 0.0f;

    const int c = ks >> 1;
    for (int i = 0; i < ks; ++i) {
        const int ny = py + (i - c) * dil;
        for (int j = 0; j < ks; ++j) {
            const int nx   = px + (j - c) * dil;
            const float bv = rpb[i * ks + j];
            const bool in  = (ny >= 0) & (ny < WDIM) & (nx >= 0) & (nx < WDIM);
            const int  np  = (ny << 6) + nx;

            float logit = bv;
            if (in) {
                float dot = 0.0f;
#pragma unroll
                for (int d = 0; d < HDIM; ++d)
                    dot = fmaf(q[d], kbp[(size_t)d * HW + np], dot);
                logit = dot + bv;
            }
            const float nm   = fmaxf(m, logit);
            const float corr = __expf(m - nm);
            const float w    = __expf(logit - nm);
            s = s * corr + w;
            if (in) {
#pragma unroll
                for (int d = 0; d < HDIM; ++d)
                    o[d] = fmaf(w, vbp[(size_t)d * HW + np], o[d] * corr);
            } else {
#pragma unroll
                for (int d = 0; d < HDIM; ++d) o[d] *= corr;
            }
            m = nm;
        }
    }

    const float inv = 1.0f / s;
    float* op = att + ((size_t)b * 256 + head * HDIM) * HW + p;
#pragma unroll
    for (int d = 0; d < HDIM; ++d) op[(size_t)d * HW] = o[d] * inv;
}

// ---------------------------------------------------------------------------
extern "C" void kernel_launch(void* const* d_in, const int* in_sizes, int n_in,
                              void* d_out, int out_size, void* d_ws, size_t ws_size,
                              hipStream_t stream)
{
    const float* x      = (const float*)d_in[0];
    const float* qkv_w  = (const float*)d_in[1];
    const float* qkv_b  = (const float*)d_in[2];
    const float* proj_w = (const float*)d_in[3];
    const float* proj_b = (const float*)d_in[4];
    const float* rpb0   = (const float*)d_in[5];
    const float* rpb1   = (const float*)d_in[6];
    const float* rpb2   = (const float*)d_in[7];
    float*       out    = (float*)d_out;

    // Workspace layout (all offsets 32B-aligned):
    char* ws = (char*)d_ws;
    float* qkv    = (float*)(ws);                                   // 4*768*4096 f32 = 48 MB
    float* att    = (float*)(ws + (size_t)48 * 1024 * 1024);        // 4*256*4096 f32 = 16 MB
    f16*   xpack  = (f16*)  (ws + (size_t)64 * 1024 * 1024);        // 4*256*4096 f16 =  8 MB
    f16*   apack  = (f16*)  (ws + (size_t)72 * 1024 * 1024);        // 4*256*4096 f16 =  8 MB
    f16*   wq_p   = (f16*)  (ws + (size_t)80 * 1024 * 1024);        // 768*256 f16 = 384 KB
    f16*   wp_p   = (f16*)  (ws + (size_t)80 * 1024 * 1024 + 512 * 1024); // 256*256 f16 = 128 KB

    dim3 blk(256);
    // Pack weights into WMMA A-fragments (once per call; tiny).
    pack_w_f16<<<dim3(48), blk, 0, stream>>>(qkv_w, wq_p);   // 48 mtiles * 8 kb * 32 lanes
    pack_w_f16<<<dim3(16), blk, 0, stream>>>(proj_w, wp_p);  // 16 mtiles
    // Pack x into WMMA B-fragments (strided gather + cvt done once).
    pack_x_f16<<<dim3(1024), blk, 0, stream>>>(x, xpack);
    // QKV projection: O = 768.
    gemm_wmma_packed<<<dim3(8, 48, 4), blk, 0, stream>>>(wq_p, qkv_b, xpack, qkv, 768);
    // Multi-range neighborhood attention.
    attn_multirange<<<dim3(512), blk, 0, stream>>>(qkv, rpb0, rpb1, rpb2, att);
    // Pack attention output for the projection GEMM.
    pack_x_f16<<<dim3(1024), blk, 0, stream>>>(att, apack);
    // Output projection: O = 256.
    gemm_wmma_packed<<<dim3(8, 16, 4), blk, 0, stream>>>(wp_p, proj_b, apack, out, 256);
}
